// GATLayer_44830868636063
// MI455X (gfx1250) — matrix operand order repaired and checked by reading, not compiled, not run on hardware
//
#include <hip/hip_runtime.h>
#include <hip/hip_bf16.h>
#include <stdint.h>

// ---------------------------------------------------------------------------
// GAT layer, N=8192, IN=256, OUT=128 on MI455X (gfx1250, wave32, WMMA).
//
// softmax(where(adj>0, a1_i + a2_j, -1e16)) row-wise: a1_i is constant along
// the softmax axis => cancels. With M = max_j a2_j:
//   w_ij = adj_ij * exp(a2_j - M);  out_i = (sum_j w_ij h_j) / (sum_j w_ij)
// Empty row => reference softmax uniform => out_i = mean_j h_j.
// The denominator sum_j w_ij is fused into the WMMA pass as an extra
// ones-column WMMA on the shared A fragment, so adj streams from HBM ONCE
// (256 MB @ 23.3 TB/s ~ 11 us lower bound).
// ---------------------------------------------------------------------------

typedef __attribute__((ext_vector_type(16))) _Float16 v16h;
typedef __attribute__((ext_vector_type(8)))  _Float16 v8h;
typedef __attribute__((ext_vector_type(8)))  float    v8f;
typedef __attribute__((ext_vector_type(4)))  int      v4i;

#define N_NODES 8192
#define IN_DIM  256
#define OUT_DIM 128

#if defined(__has_builtin)
#if __has_builtin(__builtin_amdgcn_global_load_async_to_lds_b128)
#define USE_ASYNC_B 1
#endif
#endif
#ifndef USE_ASYNC_B
#define USE_ASYNC_B 0
#endif

#if USE_ASYNC_B
// Signature leaked by clang diagnostics: (v4i AS1*, v4i AS3*, imm, imm).
typedef __attribute__((address_space(1))) v4i* gv4i_p;
typedef __attribute__((address_space(3))) v4i* lv4i_p;
#endif

union AFrag { v16h v; unsigned u[8]; };

__device__ __forceinline__ unsigned orderedKey(float f) {
  unsigned u = __float_as_uint(f);
  return (u & 0x80000000u) ? ~u : (u | 0x80000000u);
}
__device__ __forceinline__ float keyToFloat(unsigned k) {
  unsigned u = (k & 0x80000000u) ? (k & 0x7fffffffu) : ~k;
  return __uint_as_float(u);
}

// ---------------------------------------------------------------------------
__global__ void gat_init_kernel(float* colmean, unsigned* maxkey) {
  int t = threadIdx.x;
  if (t < OUT_DIM) colmean[t] = 0.0f;
  if (t == OUT_DIM) *maxkey = 0u;  // decodes to a hugely negative float
}

// ---------------------------------------------------------------------------
// h = features @ W + b (fp32), stored transposed as f16 (2 MB -> L2 resident).
// Also: colmean fallback, a2_i = h_i . a2_w + a2_b, global max(a2).
__global__ __launch_bounds__(128)
void gat_proj_kernel(const float* __restrict__ feat, const float* __restrict__ W,
                     const float* __restrict__ bias, const float* __restrict__ a2_w,
                     const float* __restrict__ a2_b,
                     _Float16* __restrict__ hT, float* __restrict__ a2,
                     float* __restrict__ colmean, unsigned* __restrict__ maxkey) {
  __shared__ float fs[IN_DIM];
  __shared__ float red[128];
  const int i = blockIdx.x;
  const int n = threadIdx.x;
  fs[n]       = feat[(size_t)i * IN_DIM + n];
  fs[n + 128] = feat[(size_t)i * IN_DIM + 128 + n];
  __syncthreads();
  float acc = bias[n];
#pragma unroll 8
  for (int k = 0; k < IN_DIM; ++k)
    acc = fmaf(fs[k], W[k * OUT_DIM + n], acc);
  hT[(size_t)n * N_NODES + i] = (_Float16)acc;
  atomicAdd(&colmean[n], acc * (1.0f / (float)N_NODES));
  red[n] = acc * a2_w[n];
  __syncthreads();
  for (int s = 64; s > 0; s >>= 1) {
    if (n < s) red[n] += red[n + s];
    __syncthreads();
  }
  if (n == 0) {
    float v = red[0] + a2_b[0];
    a2[i] = v;
    atomicMax(maxkey, orderedKey(v));
  }
}

// ---------------------------------------------------------------------------
__global__ void gat_pexp_kernel(const float* __restrict__ a2,
                                const unsigned* __restrict__ maxkey,
                                _Float16* __restrict__ p16) {
  int j = blockIdx.x * blockDim.x + threadIdx.x;
  if (j >= N_NODES) return;
  float M = keyToFloat(*maxkey);
  p16[j] = (_Float16)__expf(a2[j] - M);
}

// ---------------------------------------------------------------------------
// out = (A_w @ h) / rowsum(A_w), A_w[i][j] = adj_ij * p_j, fused denominator.
// 512 WGs x 256 threads (8 wave32): each WG owns 16 rows x 128 cols
// (1 row-tile x 8 col-tiles of 16x16); wave w owns col-tile w; wave 0 also
// multiplies the shared A frag by a ones-in-column-0 B frag -> row sums.
// Double-buffered LDS (one barrier per 32-wide K chunk), software-pipelined
// non-temporal adj loads, async-to-LDS h-tile staging.
#define LDA 40   // padded row strides (halves) -> conflict-free frag reads
#define LDB 40
__global__ __launch_bounds__(256)
void gat_attn_kernel(const int* __restrict__ adj, const _Float16* __restrict__ hT,
                     const _Float16* __restrict__ p16, const float* __restrict__ colmean,
                     float* __restrict__ out) {
  __shared__ __align__(16) _Float16 lA[2][16 * LDA];
  __shared__ __align__(16) _Float16 lB[2][128 * LDB];
  __shared__ float sInv[16];
  const int t  = threadIdx.x;
  const int i0 = blockIdx.x * 16;
  // A staging: threads 0..63, 16 rows x 32 k
  const bool aLd = (t < 64);
  const int ar   = (t >> 2) & 15;
  const int aoff = (t & 3) * 8;
  // B staging: all 256 threads, 128 rows x 32 k
  const int bn    = t >> 1;
  const int bhalf = (t & 1) * 16;
  // wave/tile
  const int w    = t >> 5;
  const int lane = t & 31;
  const int m    = lane & 15;
  const int hi   = lane >> 4;

  v8f acc  = v8f{};
  v8f accS = v8f{};
  const _Float16 zh = (_Float16)0.0f;

  // ones-in-column-0 B fragment (lane n==0 holds all-K ones)
  AFrag onesB;
#pragma unroll
  for (int g = 0; g < 8; ++g) onesB.u[g] = (m == 0) ? 0x3C003C00u : 0u;

  const int* adjRow = adj + (size_t)(i0 + ar) * N_NODES + aoff;
  const _Float16* hRow = hT + (size_t)bn * N_NODES + bhalf;

  // pipeline prologue: chunk-0 A regs
  v4i ca0{}, ca1{};
  v8h cp{};
  if (aLd) {
    const v4i* ap = (const v4i*)adjRow;
    ca0 = __builtin_nontemporal_load(ap);
    ca1 = __builtin_nontemporal_load(ap + 1);
    cp  = *(const v8h*)(p16 + aoff);
  }

  for (int chunk = 0; chunk < N_NODES / 32; ++chunk) {
    const int k0 = chunk * 32;
    const int pb = chunk & 1;

    // ---- B tile: global -> LDS (async DMA, tracked by ASYNCcnt) ----
#if USE_ASYNC_B
    {
      gv4i_p g0 = (gv4i_p)(uintptr_t)(hRow + k0);
      gv4i_p g1 = (gv4i_p)(uintptr_t)(hRow + k0 + 8);
      lv4i_p l0 = (lv4i_p)(uintptr_t)&lB[pb][bn * LDB + bhalf];
      lv4i_p l1 = (lv4i_p)(uintptr_t)&lB[pb][bn * LDB + bhalf + 8];
      __builtin_amdgcn_global_load_async_to_lds_b128(g0, l0, 0, 0);
      __builtin_amdgcn_global_load_async_to_lds_b128(g1, l1, 0, 0);
    }
#else
    const v8h* hp = (const v8h*)(hRow + k0);
    v8h h0 = hp[0];
    v8h h1 = hp[1];
#endif

    // ---- software pipeline: issue next chunk's adj/p loads ----
    v4i na0{}, na1{};
    v8h np{};
    if (aLd && chunk + 1 < N_NODES / 32) {
      const v4i* ap = (const v4i*)(adjRow + k0 + 32);
      na0 = __builtin_nontemporal_load(ap);
      na1 = __builtin_nontemporal_load(ap + 1);
      np  = *(const v8h*)(p16 + k0 + 32 + aoff);
      if (k0 + 64 < N_NODES)
        __builtin_prefetch(adjRow + k0 + 64, 0, 0);  // global_prefetch_b8
    }

    // ---- stage masked A: lA[row][k] = adj ? p16[k] : 0 ----
    if (aLd) {
      _Float16* ad = &lA[pb][ar * LDA + aoff];
      ad[0] = ca0.x > 0 ? cp[0] : zh;
      ad[1] = ca0.y > 0 ? cp[1] : zh;
      ad[2] = ca0.z > 0 ? cp[2] : zh;
      ad[3] = ca0.w > 0 ? cp[3] : zh;
      ad[4] = ca1.x > 0 ? cp[4] : zh;
      ad[5] = ca1.y > 0 ? cp[5] : zh;
      ad[6] = ca1.z > 0 ? cp[6] : zh;
      ad[7] = ca1.w > 0 ? cp[7] : zh;
    }
#if USE_ASYNC_B
#if defined(__has_builtin) && __has_builtin(__builtin_amdgcn_s_wait_asynccnt)
    __builtin_amdgcn_s_wait_asynccnt(0);
#else
    asm volatile("s_wait_asynccnt 0x0" ::: "memory");
#endif
#else
    *(v8h*)&lB[pb][bn * LDB + bhalf]     = h0;
    *(v8h*)&lB[pb][bn * LDB + bhalf + 8] = h1;
#endif
    __syncthreads();

    // ---- A fragment (16-bit A 16x32: g<4 -> K=2g+8hi, else 16+2(g-4)+8hi) ----
    const unsigned* lAu = (const unsigned*)lA[pb];
    const unsigned* lBu = (const unsigned*)lB[pb];
    AFrag af;
    const int arow = m * (LDA / 2);
#pragma unroll
    for (int g = 0; g < 8; ++g) {
      int K = (g < 4) ? (2 * g + 8 * hi) : (16 + 2 * (g - 4) + 8 * hi);
      af.u[g] = lAu[arow + (K >> 1)];
    }
    // ---- B fragment (col-tile w; lane indexes N, K same as A frag) ----
    AFrag bf;
    const int brow = (16 * w + m) * (LDB / 2);
#pragma unroll
    for (int g = 0; g < 8; ++g) bf.u[g] = lBu[brow + ((g < 4) ? (4 * hi + g) : (8 + 4 * hi + (g - 4)))];

    acc = __builtin_amdgcn_wmma_f32_16x16x32_f16(
        false, af.v, false, bf.v, (short)0, acc, false, false);
    if (w == 0)  // fused row-sum: same A frag x ones-column => denominator
      accS = __builtin_amdgcn_wmma_f32_16x16x32_f16(
          false, af.v, false, onesB.v, (short)0, accS, false, false);

    ca0 = na0; ca1 = na1; cp = np;
  }

  // ---- publish per-row 1/s (s in column 0 of accS: lanes 0 and 16) ----
  if (w == 0 && m == 0) {
#pragma unroll
    for (int v = 0; v < 8; ++v) {
      float sv = accS[v];
      sInv[v + 8 * hi] = (sv > 0.0f) ? (1.0f / sv) : -1.0f;  // -1 => empty row
    }
  }
  __syncthreads();

  // ---- epilogue: C layout VGPR v, lane: M = v + 8hi, N = lane&15 ----
  const int col = 16 * w + m;
  const float cm = colmean[col];
#pragma unroll
  for (int v = 0; v < 8; ++v) {
    const int i = i0 + v + 8 * hi;
    const float inv = sInv[v + 8 * hi];
    out[(size_t)i * OUT_DIM + col] = (inv > 0.0f) ? (acc[v] * inv) : cm;
  }
}

// ---------------------------------------------------------------------------
extern "C" void kernel_launch(void* const* d_in, const int* in_sizes, int n_in,
                              void* d_out, int out_size, void* d_ws, size_t ws_size,
                              hipStream_t stream) {
  (void)in_sizes; (void)n_in; (void)out_size; (void)ws_size;
  const float* feat = (const float*)d_in[0];   // [8192,256]
  const int*   adj  = (const int*)d_in[1];     // [8192,8192]
  const float* W    = (const float*)d_in[2];   // [256,128]
  const float* bias = (const float*)d_in[3];   // [128]
  // d_in[4] (a1_w), d_in[5] (a1_b) unused: a1_i cancels in the row softmax.
  const float* a2w  = (const float*)d_in[6];   // [128]
  const float* a2b  = (const float*)d_in[7];   // [1]
  float* out = (float*)d_out;                  // [8192,128]

  char* ws = (char*)d_ws;
  _Float16* hT      = (_Float16*)(ws);                    // 2,097,152 B
  float*    a2      = (float*)(ws + 2097152);             //    32,768 B
  _Float16* p16     = (_Float16*)(ws + 2097152 + 32768);  //    16,384 B
  float*    colmean = (float*)(ws + 2097152 + 49152);     //       512 B
  unsigned* maxkey  = (unsigned*)(ws + 2097152 + 49664);  //         4 B

  gat_init_kernel<<<1, 256, 0, stream>>>(colmean, maxkey);
  gat_proj_kernel<<<N_NODES, 128, 0, stream>>>(feat, W, bias, a2w, a2b,
                                               hT, a2, colmean, maxkey);
  gat_pexp_kernel<<<N_NODES / 256, 256, 0, stream>>>(a2, maxkey, p16);
  gat_attn_kernel<<<N_NODES / 16, 256, 0, stream>>>(adj, hT, p16, colmean, out);
}
